// ConvPosMultiHeadAttn_Order_1786706395835
// MI455X (gfx1250) — matrix operand order repaired and checked by reading, not compiled, not run on hardware
//
#include <hip/hip_runtime.h>
#include <cstdint>
#include <cstddef>

// ---------------------------------------------------------------------------
// Types for CDNA5 WMMA (wave32): v_wmma_f32_16x16x32_bf16
// ---------------------------------------------------------------------------
typedef __attribute__((ext_vector_type(16))) __bf16 v16bf;
typedef __attribute__((ext_vector_type(8)))  float  v8f;
typedef __attribute__((ext_vector_type(4)))  unsigned int v4u;

union Frag16 {
    v16bf bf;
    v4u   q[2];
};

__device__ __forceinline__ v8f wmma_bf16(v16bf a, v16bf b, v8f c) {
    // 8 args: (neg_a, A, neg_b, B, c_mod, C, reuse_a, reuse_b)
    return __builtin_amdgcn_wmma_f32_16x16x32_bf16(false, a, false, b,
                                                   (short)0, c, false, false);
}

__device__ __forceinline__ unsigned short f2bf(float x) {
    unsigned int u = __float_as_uint(x);
    u += 0x7FFFu + ((u >> 16) & 1u);          // round-to-nearest-even
    return (unsigned short)(u >> 16);
}

// ---------------------------------------------------------------------------
// CDNA5 async copy: GLOBAL_LOAD_ASYNC_TO_LDS_B128 (ASYNCcnt-tracked)
// Per-lane: 16 bytes  global[vaddr]  ->  LDS[vdst]
// ---------------------------------------------------------------------------
__device__ __forceinline__ void async_load_b128(const unsigned short* gptr,
                                                unsigned int lds_off) {
    asm volatile("global_load_async_to_lds_b128 %0, %1, off"
                 :: "v"(lds_off), "v"(gptr)
                 : "memory");
}

template <int N>
__device__ __forceinline__ void wait_asynccnt() {
#if __has_builtin(__builtin_amdgcn_s_wait_asynccnt)
    __builtin_amdgcn_s_wait_asynccnt(N);
#else
    asm volatile("s_wait_asynccnt %0" :: "n"(N) : "memory");
#endif
}

// Generic-LDS pointer -> LDS byte offset (low 32 bits of the address)
__device__ __forceinline__ unsigned int lds_off_of(const void* p) {
    return (unsigned int)(size_t)p;
}

// A-operand fragment (16x32 bf16, MxK), row-major source, per ISA 7.12.2:
// lanes 0-15: row m=lane, halves = K[kb+0..7] then K[kb+16..23]
// lanes 16-31: row m=lane-16, halves = K[kb+8..15] then K[kb+24..31]
__device__ __forceinline__ v16bf load_a_frag(const unsigned short* __restrict__ base,
                                             int row0, int ld, int kb, int lane) {
    int m  = row0 + (lane & 15);
    int ko = (lane & 16) ? 8 : 0;
    const unsigned short* p = base + (size_t)m * ld + kb + ko;
    Frag16 f;
    f.q[0] = *(const v4u*)(p);
    f.q[1] = *(const v4u*)(p + 16);
    return f.bf;
}

// B-operand fragment (32x16 bf16, KxN) sourced from row-major Bt[N][K]
// (B[k][n] = Bt[n][k]); lanes 0-15: col n=lane, K[kb+0..7] then K[kb+16..23];
// lanes 16-31: K[kb+8..15] then K[kb+24..31].
__device__ __forceinline__ v16bf load_b_frag(const unsigned short* __restrict__ base,
                                             int row0, int ld, int kb, int lane) {
    int n  = row0 + (lane & 15);
    int ko = (lane & 16) ? 8 : 0;
    const unsigned short* p = base + (size_t)n * ld + kb + ko;
    Frag16 f;
    f.q[0] = *(const v4u*)(p);
    f.q[1] = *(const v4u*)(p + 16);
    return f.bf;
}

// ---------------------------------------------------------------------------
// Async-staged NT-GEMM mainloop. Block tile M=128, N=64, Kstep=32.
// 256 threads issue 3 async b128 each per stage (A: 2 x 64 rows, B: 64 rows).
// Double-buffered LDS; per-wave fragment reads via ds_load_b128.
// ---------------------------------------------------------------------------
__device__ __forceinline__ void async_stage(const unsigned short* __restrict__ A,
                                            const unsigned short* __restrict__ Bt,
                                            int lda, int ldb, int mblk, int nblk, int kb,
                                            unsigned short* Ab, unsigned short* Bb, int t) {
    int r   = t >> 2;
    int seg = (t & 3) * 8;
    async_load_b128(A + (size_t)(mblk + r) * lda + kb + seg,
                    lds_off_of(Ab + r * 32 + seg));
    async_load_b128(A + (size_t)(mblk + 64 + r) * lda + kb + seg,
                    lds_off_of(Ab + (64 + r) * 32 + seg));
    async_load_b128(Bt + (size_t)(nblk + r) * ldb + kb + seg,
                    lds_off_of(Bb + r * 32 + seg));
}

__device__ __forceinline__ void gemm_mainloop_lds(const unsigned short* __restrict__ A,
                                                  const unsigned short* __restrict__ Bt,
                                                  int lda, int ldb, int K,
                                                  int mblk, int nblk,
                                                  unsigned short* AbufBase,  // [2][128*32]
                                                  unsigned short* BbufBase,  // [2][64*32]
                                                  int m0l, int n0l, int lane, int t,
                                                  v8f c[2][2]) {
    async_stage(A, Bt, lda, ldb, mblk, nblk, 0, AbufBase, BbufBase, t);
    int nk = K >> 5;
    for (int i = 0; i < nk; i++) {
        int cur = i & 1;
        unsigned short* Ab = AbufBase + cur * (128 * 32);
        unsigned short* Bb = BbufBase + cur * (64 * 32);
        if (i + 1 < nk) {
            async_stage(A, Bt, lda, ldb, mblk, nblk, (i + 1) << 5,
                        AbufBase + (cur ^ 1) * (128 * 32),
                        BbufBase + (cur ^ 1) * (64 * 32), t);
            wait_asynccnt<3>();     // prefetch stage still in flight
        } else {
            wait_asynccnt<0>();
        }
        __syncthreads();            // all waves' async data visible
        v16bf a0 = load_a_frag(Ab, m0l,      32, 0, lane);
        v16bf a1 = load_a_frag(Ab, m0l + 16, 32, 0, lane);
        v16bf b0 = load_b_frag(Bb, n0l,      32, 0, lane);
        v16bf b1 = load_b_frag(Bb, n0l + 16, 32, 0, lane);
        c[0][0] = wmma_bf16(a0, b0, c[0][0]);
        c[0][1] = wmma_bf16(a0, b1, c[0][1]);
        c[1][0] = wmma_bf16(a1, b0, c[1][0]);
        c[1][1] = wmma_bf16(a1, b1, c[1][1]);
        __syncthreads();            // done reading before buffer is overwritten
    }
}

#define GEMM_PROLOGUE()                                                     \
    __shared__ __align__(16) unsigned short Abuf[2][128 * 32];              \
    __shared__ __align__(16) unsigned short Bbuf[2][64 * 32];               \
    int t = threadIdx.x;                                                    \
    int lane = t & 31, wid = t >> 5;                                        \
    int mblk = blockIdx.y * 128, nblk = blockIdx.x * 64;                    \
    int m0l = (wid & 3) * 32, n0l = (wid >> 2) * 32;                        \
    int m0 = mblk + m0l, n0 = nblk + n0l;                                   \
    (void)m0; (void)n0;

// ---------------------------------------------------------------------------
// Elementwise prep kernels
// ---------------------------------------------------------------------------
__global__ void cvt_bf16_kernel(const float* __restrict__ src,
                                unsigned short* __restrict__ dst, int n) {
    int i = blockIdx.x * blockDim.x + threadIdx.x;
    if (i < n) dst[i] = f2bf(src[i]);
}

// Wt[n*K + k] = bf16(W[k*N + n])
__global__ void transpose_bf16_kernel(const float* __restrict__ W,
                                      unsigned short* __restrict__ Wt,
                                      int K, int N) {
    int i = blockIdx.x * blockDim.x + threadIdx.x;
    if (i >= K * N) return;
    int n = i / K, k = i % K;
    Wt[i] = f2bf(W[(size_t)k * N + n]);
}

// Relative sinusoidal PE, rows for positions l-512, l in [0,1024)
__global__ void pe_kernel(unsigned short* __restrict__ pe) {
    int i = blockIdx.x * blockDim.x + threadIdx.x;
    if (i >= 1024 * 64) return;
    int l = i >> 6, j = i & 63;
    float p    = (float)(l - 512);
    float freq = __expf(-(float)(j & 31) * 0.2971077539347156f); // ln(1e4)/31
    float ang  = p * freq;
    float v    = (j < 32) ? __sinf(ang) : __cosf(ang);
    pe[i] = f2bf(v);
}

// ---------------------------------------------------------------------------
// GEMM 1: QKV = X[4096,1024] * WqkvT[4096,1024]^T; scatter per-head bf16.
// ---------------------------------------------------------------------------
__global__ __launch_bounds__(256) void gemm_qkv_kernel(
        const unsigned short* __restrict__ X, const unsigned short* __restrict__ Wt,
        unsigned short* __restrict__ Qb,  unsigned short* __restrict__ K1b,
        unsigned short* __restrict__ K2b, unsigned short* __restrict__ Vt) {
    GEMM_PROLOGUE();
    v8f c[2][2] = {};
    gemm_mainloop_lds(X, Wt, 1024, 1024, 1024, mblk, nblk,
                      &Abuf[0][0], &Bbuf[0][0], m0l, n0l, lane, t, c);
    int half8 = (lane & 16) ? 8 : 0;
    for (int mi = 0; mi < 2; mi++)
        for (int ni = 0; ni < 2; ni++)
            for (int v = 0; v < 8; v++) {
                int m = m0 + mi * 16 + v + half8;
                int n = n0 + ni * 16 + (lane & 15);
                unsigned short bv = f2bf(c[mi][ni][v]);
                int b = m >> 10, q = m & 1023;
                int sec = n >> 10, h = (n & 1023) >> 6, d = n & 63;
                size_t bh = (size_t)(b * 16 + h);
                if (sec == 0)      Qb [(bh * 1024 + q) * 64 + d] = bv;
                else if (sec == 1) K1b[(bh * 1024 + q) * 64 + d] = bv;
                else if (sec == 2) K2b[(bh * 1024 + q) * 64 + d] = bv;
                else               Vt [(bh * 64 + d) * 1024 + q] = bv;   // transposed V
            }
}

// ---------------------------------------------------------------------------
// GEMM 2: POS = pe[1024,64] * WposT[3072,64]^T; scatter qp/kp1/kp2 bf16.
// ---------------------------------------------------------------------------
__global__ __launch_bounds__(256) void gemm_pos_kernel(
        const unsigned short* __restrict__ PE, const unsigned short* __restrict__ Wt,
        unsigned short* __restrict__ QPb, unsigned short* __restrict__ KP1b,
        unsigned short* __restrict__ KP2b) {
    GEMM_PROLOGUE();
    v8f c[2][2] = {};
    gemm_mainloop_lds(PE, Wt, 64, 64, 64, mblk, nblk,
                      &Abuf[0][0], &Bbuf[0][0], m0l, n0l, lane, t, c);
    int half8 = (lane & 16) ? 8 : 0;
    for (int mi = 0; mi < 2; mi++)
        for (int ni = 0; ni < 2; ni++)
            for (int v = 0; v < 8; v++) {
                int m = m0 + mi * 16 + v + half8;           // position l
                int n = n0 + ni * 16 + (lane & 15);
                unsigned short bv = f2bf(c[mi][ni][v]);
                int sec = n >> 10, h = (n & 1023) >> 6, d = n & 63;
                size_t idx = ((size_t)h * 1024 + m) * 64 + d;
                if (sec == 0)      QPb [idx] = bv;
                else if (sec == 1) KP1b[idx] = bv;
                else               KP2b[idx] = bv;
            }
}

// ---------------------------------------------------------------------------
// Fused attention (flash-style, one 16-row q tile per wave).
// All 8 waves of a block share qt -> uniform trip count -> barriers legal.
// ---------------------------------------------------------------------------
__global__ __launch_bounds__(256) void attn_kernel(
        const unsigned short* __restrict__ Qb,  const unsigned short* __restrict__ K1b,
        const unsigned short* __restrict__ K2b, const unsigned short* __restrict__ Vt,
        const unsigned short* __restrict__ QPb, const unsigned short* __restrict__ KP1b,
        const unsigned short* __restrict__ KP2b,
        const int* __restrict__ qmask, const int* __restrict__ umask,
        unsigned short* __restrict__ Obf) {
    __shared__ __align__(16) unsigned short plds[8][512];   // 16x32 bf16 per wave
    int lane = threadIdx.x & 31, wid = threadIdx.x >> 5;
    int qt = blockIdx.x >> 3;
    int bh = (blockIdx.x & 7) * 8 + wid;
    int b = bh >> 4, h = bh & 15;
    int q0 = qt * 16;

    const unsigned short* Q   = Qb  + (size_t)bh * 1024 * 64;
    const unsigned short* K1  = K1b + (size_t)bh * 1024 * 64;
    const unsigned short* K2  = K2b + (size_t)bh * 1024 * 64;
    const unsigned short* V   = Vt  + (size_t)bh * 64 * 1024;
    const unsigned short* QP  = QPb  + (size_t)h * 1024 * 64;
    const unsigned short* KP1 = KP1b + (size_t)h * 1024 * 64;
    const unsigned short* KP2 = KP2b + (size_t)h * 1024 * 64;
    const int* qm = qmask + (size_t)b * 1024;
    const int* um = umask + (size_t)b * 1024;

    v16bf qa0 = load_a_frag(Q,  q0, 64, 0,  lane);
    v16bf qa1 = load_a_frag(Q,  q0, 64, 32, lane);
    v16bf pa0 = load_a_frag(QP, q0, 64, 0,  lane);
    v16bf pa1 = load_a_frag(QP, q0, 64, 32, lane);

    int half8 = (lane & 16) ? 8 : 0;
    int qmq[8], qrow[8];
    for (int v = 0; v < 8; v++) { qrow[v] = q0 + v + half8; qmq[v] = qm[qrow[v]]; }

    float rmax[8], rsum[8];
    v8f o[4] = {};
    for (int v = 0; v < 8; v++) { rmax[v] = -1e30f; rsum[v] = 0.f; }

    int nkb = (q0 + 47) >> 5;                      // k-blocks of 32, causal bound
    for (int kb = 0; kb < nkb; kb++) {
        int k0 = kb * 32;
        v8f s1[2] = {{}, {}}, s2[2] = {{}, {}};
        for (int nt = 0; nt < 2; nt++) {
            int kr = k0 + nt * 16;
            // S1 = q*k1^T + qp*kp1^T  (K concat = 128)
            s1[nt] = wmma_bf16(qa0, load_b_frag(K1,  kr, 64, 0,  lane), s1[nt]);
            s1[nt] = wmma_bf16(qa1, load_b_frag(K1,  kr, 64, 32, lane), s1[nt]);
            s1[nt] = wmma_bf16(pa0, load_b_frag(KP1, kr, 64, 0,  lane), s1[nt]);
            s1[nt] = wmma_bf16(pa1, load_b_frag(KP1, kr, 64, 32, lane), s1[nt]);
            // S2 = q*k2^T + qp*kp2^T
            s2[nt] = wmma_bf16(qa0, load_b_frag(K2,  kr, 64, 0,  lane), s2[nt]);
            s2[nt] = wmma_bf16(qa1, load_b_frag(K2,  kr, 64, 32, lane), s2[nt]);
            s2[nt] = wmma_bf16(pa0, load_b_frag(KP2, kr, 64, 0,  lane), s2[nt]);
            s2[nt] = wmma_bf16(pa1, load_b_frag(KP2, kr, 64, 32, lane), s2[nt]);
        }

        // speaker select + causal/pad mask in C-layout, per-row tile max
        float sv[2][8], tmax[8];
        for (int v = 0; v < 8; v++) tmax[v] = -1e30f;
        for (int nt = 0; nt < 2; nt++) {
            int kc  = k0 + nt * 16 + (lane & 15);
            int qmk = qm[kc];
            int uok = um[kc];
            for (int v = 0; v < 8; v++) {
                float s = (qmq[v] == qmk) ? s1[nt][v] : s2[nt][v];
                bool valid = (kc <= qrow[v]) && (uok != 0);
                s = valid ? s : -1e30f;
                sv[nt][v] = s;
                tmax[v] = fmaxf(tmax[v], s);
            }
        }
        for (int v = 0; v < 8; v++) {
            float tm = tmax[v];
            tm = fmaxf(tm, __shfl_xor(tm, 1));
            tm = fmaxf(tm, __shfl_xor(tm, 2));
            tm = fmaxf(tm, __shfl_xor(tm, 4));
            tm = fmaxf(tm, __shfl_xor(tm, 8));
            tmax[v] = tm;
        }
        float scale[8];
        for (int v = 0; v < 8; v++) {
            float nm = fmaxf(rmax[v], tmax[v]);
            scale[v] = __expf(rmax[v] - nm);
            rmax[v] = nm;
        }
        float tsum[8];
        for (int v = 0; v < 8; v++) tsum[v] = 0.f;
        for (int nt = 0; nt < 2; nt++)
            for (int v = 0; v < 8; v++) {
                float p = __expf(sv[nt][v] - rmax[v]);
                sv[nt][v] = p;
                tsum[v] += p;
            }
        for (int v = 0; v < 8; v++) {
            float ts = tsum[v];
            ts += __shfl_xor(ts, 1);
            ts += __shfl_xor(ts, 2);
            ts += __shfl_xor(ts, 4);
            ts += __shfl_xor(ts, 8);
            rsum[v] = rsum[v] * scale[v] + ts;
        }
        for (int dt = 0; dt < 4; dt++)
            for (int v = 0; v < 8; v++) o[dt][v] *= scale[v];

        // relayout P: C-layout fp32 -> A-layout bf16 via per-wave LDS bounce
        unsigned short* pl = plds[wid];
        __syncthreads();                          // WAR vs previous read
        for (int nt = 0; nt < 2; nt++) {
            int col = nt * 16 + (lane & 15);
            for (int v = 0; v < 8; v++)
                pl[(v + half8) * 32 + col] = f2bf(sv[nt][v]);
        }
        __syncthreads();                          // RAW
        v16bf pA;
        {
            int m  = lane & 15;
            int ko = (lane & 16) ? 8 : 0;
            const unsigned short* p = pl + m * 32 + ko;
            Frag16 f;
            f.q[0] = *(const v4u*)(p);
            f.q[1] = *(const v4u*)(p + 16);
            pA = f.bf;
        }
        // O += P * V   (B operand from transposed V -> contiguous rows)
        for (int dt = 0; dt < 4; dt++) {
            v16bf vb = load_b_frag(V, dt * 16, 1024, k0, lane);
            o[dt] = wmma_bf16(pA, vb, o[dt]);
        }
    }

    // epilogue: O /= rowsum, pack heads -> [B*L, 1024] bf16
    for (int dt = 0; dt < 4; dt++)
        for (int v = 0; v < 8; v++) {
            int m = qrow[v];
            int d = dt * 16 + (lane & 15);
            float val = o[dt][v] / rsum[v];
            Obf[((size_t)b * 1024 + m) * 1024 + h * 64 + d] = f2bf(val);
        }
}

// ---------------------------------------------------------------------------
// GEMM 3: out = attn[4096,1024] * WfcT[1024,1024]^T  (fp32 output)
// ---------------------------------------------------------------------------
__global__ __launch_bounds__(256) void gemm_fc_kernel(
        const unsigned short* __restrict__ A, const unsigned short* __restrict__ Wt,
        float* __restrict__ out) {
    GEMM_PROLOGUE();
    v8f c[2][2] = {};
    gemm_mainloop_lds(A, Wt, 1024, 1024, 1024, mblk, nblk,
                      &Abuf[0][0], &Bbuf[0][0], m0l, n0l, lane, t, c);
    int half8 = (lane & 16) ? 8 : 0;
    for (int mi = 0; mi < 2; mi++)
        for (int ni = 0; ni < 2; ni++)
            for (int v = 0; v < 8; v++) {
                int m = m0 + mi * 16 + v + half8;
                int n = n0 + ni * 16 + (lane & 15);
                out[(size_t)m * 1024 + n] = c[mi][ni][v];
            }
}

// ---------------------------------------------------------------------------
// Host launcher
// ---------------------------------------------------------------------------
extern "C" void kernel_launch(void* const* d_in, const int* in_sizes, int n_in,
                              void* d_out, int out_size, void* d_ws, size_t ws_size,
                              hipStream_t stream) {
    const float* embed = (const float*)d_in[0];   // [4,1024,1024]
    const int*   umask = (const int*)d_in[1];     // [4,1024]
    const int*   qmask = (const int*)d_in[2];     // [4,1024]
    const float* W_qkv = (const float*)d_in[3];   // [1024,4096]
    const float* W_pos = (const float*)d_in[4];   // [64,3072]
    const float* W_fc  = (const float*)d_in[5];   // [1024,1024]
    (void)in_sizes; (void)n_in; (void)out_size; (void)ws_size;

    unsigned char* ws = (unsigned char*)d_ws;
    size_t off = 0;
    auto alloc = [&](size_t elems) {
        void* p = ws + off;
        off = (off + elems * 2 + 255) & ~(size_t)255;
        return (unsigned short*)p;
    };
    unsigned short* Xbf    = alloc((size_t)4096 * 1024);
    unsigned short* WqkvT  = alloc((size_t)4096 * 1024);
    unsigned short* WposT  = alloc((size_t)3072 * 64);
    unsigned short* WfcT   = alloc((size_t)1024 * 1024);
    unsigned short* PEbf   = alloc((size_t)1024 * 64);
    unsigned short* Qb     = alloc((size_t)64 * 1024 * 64);
    unsigned short* K1b    = alloc((size_t)64 * 1024 * 64);
    unsigned short* K2b    = alloc((size_t)64 * 1024 * 64);
    unsigned short* Vtb    = alloc((size_t)64 * 64 * 1024);
    unsigned short* QPb    = alloc((size_t)16 * 1024 * 64);
    unsigned short* KP1b   = alloc((size_t)16 * 1024 * 64);
    unsigned short* KP2b   = alloc((size_t)16 * 1024 * 64);
    unsigned short* Attnbf = alloc((size_t)4096 * 1024);

    // Phase 1: precision prep
    cvt_bf16_kernel<<<(4096 * 1024 + 255) / 256, 256, 0, stream>>>(embed, Xbf, 4096 * 1024);
    transpose_bf16_kernel<<<(1024 * 4096 + 255) / 256, 256, 0, stream>>>(W_qkv, WqkvT, 1024, 4096);
    transpose_bf16_kernel<<<(64 * 3072 + 255) / 256, 256, 0, stream>>>(W_pos, WposT, 64, 3072);
    transpose_bf16_kernel<<<(1024 * 1024 + 255) / 256, 256, 0, stream>>>(W_fc, WfcT, 1024, 1024);
    pe_kernel<<<(1024 * 64 + 255) / 256, 256, 0, stream>>>(PEbf);

    // Phase 2: projections (WMMA + async-to-LDS double buffering)
    gemm_qkv_kernel<<<dim3(4096 / 64, 4096 / 128), 256, 0, stream>>>(Xbf, WqkvT, Qb, K1b, K2b, Vtb);
    gemm_pos_kernel<<<dim3(3072 / 64, 1024 / 128), 256, 0, stream>>>(PEbf, WposT, QPb, KP1b, KP2b);

    // Phase 3: fused attention (WMMA flash)
    attn_kernel<<<64 * 8, 256, 0, stream>>>(Qb, K1b, K2b, Vtb, QPb, KP1b, KP2b,
                                            qmask, umask, Attnbf);

    // Phase 4: output projection (WMMA, fp32 out)
    gemm_fc_kernel<<<dim3(1024 / 64, 4096 / 128), 256, 0, stream>>>(Attnbf, WfcT, (float*)d_out);
}
